// NTXentLoss_9706626090135
// MI455X (gfx1250) — compile-verified, matrix operand order checked by reading, not run on hardware
//
#include <hip/hip_runtime.h>
#include <hip/hip_bf16.h>

typedef _Float16 f16;
typedef __attribute__((ext_vector_type(16))) _Float16 v16h;
typedef __attribute__((ext_vector_type(8)))  _Float16 v8h;
typedef __attribute__((ext_vector_type(8)))  float    v8f;

#define DIM 256
#define KSTEPS 8              // 256 / 32
#define LDS_STRIDE 264        // 132 dwords, stride % 64 banks == 4 -> conflict-free b128 reads
#define INV_T 2.0f            // 1 / TEMPERATURE
#define SHIFT 2.0f            // fixed logsumexp shift (sim <= 1/T = 2)

// ---------------------------------------------------------------------------
// Kernel 1: L2-normalize rows of [z_i; z_j] and emit f16 reps [2N, 256].
// One wave (32 lanes) per row; 8 floats per lane.
// ---------------------------------------------------------------------------
__launch_bounds__(256)
__global__ void ntxent_normalize(const float* __restrict__ zi,
                                 const float* __restrict__ zj,
                                 f16* __restrict__ reps, int N) {
  const int wave = (blockIdx.x * blockDim.x + threadIdx.x) >> 5;
  const int lane = threadIdx.x & 31;
  const int row  = wave;
  if (row >= 2 * N) return;
  const float* src = (row < N) ? (zi + (size_t)row * DIM)
                               : (zj + (size_t)(row - N) * DIM);
  float4 x0 = *(const float4*)(src + lane * 8);
  float4 x1 = *(const float4*)(src + lane * 8 + 4);
  float ss = x0.x * x0.x + x0.y * x0.y + x0.z * x0.z + x0.w * x0.w +
             x1.x * x1.x + x1.y * x1.y + x1.z * x1.z + x1.w * x1.w;
#pragma unroll
  for (int m = 1; m < 32; m <<= 1) ss += __shfl_xor(ss, m, 32);
  const float inv = 1.0f / fmaxf(sqrtf(ss), 1e-12f);
  v8h o;
  o[0] = (f16)(x0.x * inv); o[1] = (f16)(x0.y * inv);
  o[2] = (f16)(x0.z * inv); o[3] = (f16)(x0.w * inv);
  o[4] = (f16)(x1.x * inv); o[5] = (f16)(x1.y * inv);
  o[6] = (f16)(x1.z * inv); o[7] = (f16)(x1.w * inv);
  *(v8h*)(reps + (size_t)row * DIM + lane * 8) = o;
}

// ---------------------------------------------------------------------------
// Kernel 2: fused sim = (R R^T)/T tiles + masked logsumexp + positive pick.
// Grid: 2N/128 workgroups of 256 threads (8 waves). Wave w owns row tile
// [base_m, base_m+16). Loops over all 2N/16 column tiles with DOUBLE-BUFFERED
// LDS staging: tile t+1 is global-loaded into registers before the WMMA chain
// of tile t and stored into the alternate buffer afterwards, hiding L2
// latency behind the 8 v_wmma ops. One barrier per iteration. Prefetch is
// unconditional (wrapped index) to keep the loop branch-free except for the
// scalar special-tile test (tdiag/tpos forced into SGPRs via readfirstlane).
// rowvals[row] = -pos(row) + logsumexp_row.
// ---------------------------------------------------------------------------
__launch_bounds__(256)
__global__ void ntxent_main(const f16* __restrict__ reps,
                            float* __restrict__ rowvals, int N2) {
  __shared__ f16 lds[2][16 * LDS_STRIDE];

  const int tid    = threadIdx.x;
  const int wave   = tid >> 5;
  const int lane   = tid & 31;
  const int hh     = lane >> 4;   // K-half selector (A/B fragment layout)
  const int l16    = lane & 15;
  const int base_m = blockIdx.x * 128 + wave * 16;
  const int myrow  = base_m + l16;
  const int N      = N2 >> 1;

  // Wave-uniform: the only tiles where diagonal / positive columns can occur.
  // Force into SGPRs so the per-tile test is a scalar branch, not exec-masking.
  const int tdiag = __builtin_amdgcn_readfirstlane(base_m >> 4);
  int posbase = base_m + N; if (posbase >= N2) posbase -= N2;
  const int tpos = __builtin_amdgcn_readfirstlane(posbase >> 4);

  // Preload A fragments for this wave's 16 rows, all K (resident in VGPRs).
  // Lane layout per ISA: half 0 lanes hold K {0..7, 16..23} of each 32-chunk,
  // half 1 lanes hold K {8..15, 24..31}.
  v16h a[KSTEPS];
#pragma unroll
  for (int kk = 0; kk < KSTEPS; ++kk) {
    const int kb = kk * 32 + hh * 8;
    v8h lo = *(const v8h*)(reps + (size_t)myrow * DIM + kb);
    v8h hi = *(const v8h*)(reps + (size_t)myrow * DIM + kb + 16);
    a[kk] = __builtin_shufflevector(lo, hi, 0, 1, 2, 3, 4, 5, 6, 7,
                                    8, 9, 10, 11, 12, 13, 14, 15);
  }

  v8f sume = {};
  v8f posv = {};

  const int nTiles = N2 >> 4;
  // Staging assignment: thread t copies one 32B chunk (row = t/16, chunk = t%16).
  const int sr = tid >> 4;
  const int sc = (tid & 15) * 16;

  // Prologue: stage tile 0 into buffer 0.
  v8h sb0, sb1;
  {
    const f16* g = reps + (size_t)sr * DIM + sc;
    sb0 = *(const v8h*)g;
    sb1 = *(const v8h*)(g + 8);
    *(v8h*)(&lds[0][sr * LDS_STRIDE + sc])     = sb0;
    *(v8h*)(&lds[0][sr * LDS_STRIDE + sc + 8]) = sb1;
  }
  int cur = 0;

  for (int t = 0; t < nTiles; ++t) {
    __syncthreads();  // buffer `cur` staged & prior reads of `cur^1` retired

    // Unconditional prefetch of the next tile (wrapped; last iter is a
    // harmless redundant stage into the never-again-read buffer).
    {
      int tn = t + 1; if (tn == nTiles) tn = 0;
      const f16* g = reps + (size_t)(tn * 16 + sr) * DIM + sc;
      sb0 = *(const v8h*)g;
      sb1 = *(const v8h*)(g + 8);
    }

    v8f c = {};
#pragma unroll
    for (int kk = 0; kk < KSTEPS; ++kk) {
      const int kb = kk * 32 + hh * 8;
      const f16* lp = &lds[cur][l16 * LDS_STRIDE + kb];
      v8h lo = *(const v8h*)lp;
      v8h hi = *(const v8h*)(lp + 16);
      v16h b = __builtin_shufflevector(lo, hi, 0, 1, 2, 3, 4, 5, 6, 7,
                                       8, 9, 10, 11, 12, 13, 14, 15);
      c = __builtin_amdgcn_wmma_f32_16x16x32_f16(
              false, a[kk], false, b, (short)0, c, false, false);
    }

    {   // store prefetched tile into the alternate buffer
      f16* buf = lds[cur ^ 1];
      *(v8h*)(&buf[sr * LDS_STRIDE + sc])     = sb0;
      *(v8h*)(&buf[sr * LDS_STRIDE + sc + 8]) = sb1;
    }

    // C layout: element (VGPR r, lane L) = (row = base_m + r + (L/16)*8,
    //                                       col = t*16 + L%16)
    if (t == tdiag || t == tpos) {
      // Rare path (<=2 of nTiles): diagonal mask + positive extraction.
      const int col = t * 16 + l16;
#pragma unroll
      for (int r = 0; r < 8; ++r) {
        const float s   = c[r] * INV_T;
        const int   row = base_m + r + hh * 8;
        int pcol = row + N; if (pcol >= N2) pcol -= N2;
        const float e = __expf(s - SHIFT);
        sume[r] += (col == row)  ? 0.0f : e;
        posv[r] += (col == pcol) ? s    : 0.0f;
      }
    } else {
      // Hot path: plain exp-accumulate, co-executes with WMMA (TRANS class).
#pragma unroll
      for (int r = 0; r < 8; ++r) {
        sume[r] += __expf(c[r] * INV_T - SHIFT);
      }
    }
    cur ^= 1;
  }

  // Reduce across the 16 lanes of each half (cols of the tile).
#pragma unroll
  for (int m = 1; m <= 8; m <<= 1) {
#pragma unroll
    for (int r = 0; r < 8; ++r) {
      sume[r] += __shfl_xor(sume[r], m, 32);
      posv[r] += __shfl_xor(posv[r], m, 32);
    }
  }

  if (l16 == 0) {
#pragma unroll
    for (int r = 0; r < 8; ++r) {
      const int row = base_m + r + hh * 8;
      rowvals[row] = -posv[r] + SHIFT + __logf(sume[r]);
    }
  }
}

// ---------------------------------------------------------------------------
// Kernel 3: deterministic mean over rowvals -> single scalar loss.
// ---------------------------------------------------------------------------
__launch_bounds__(256)
__global__ void ntxent_reduce(const float* __restrict__ rowvals,
                              float* __restrict__ out, int n) {
  __shared__ float sdata[256];
  float s = 0.0f;
  for (int i = threadIdx.x; i < n; i += 256) s += rowvals[i];
  sdata[threadIdx.x] = s;
  __syncthreads();
#pragma unroll
  for (int ofs = 128; ofs > 0; ofs >>= 1) {
    if (threadIdx.x < ofs) sdata[threadIdx.x] += sdata[threadIdx.x + ofs];
    __syncthreads();
  }
  if (threadIdx.x == 0) out[0] = sdata[0] / (float)n;
}

extern "C" void kernel_launch(void* const* d_in, const int* in_sizes, int n_in,
                              void* d_out, int out_size, void* d_ws, size_t ws_size,
                              hipStream_t stream) {
  const float* zi = (const float*)d_in[0];
  const float* zj = (const float*)d_in[1];
  const int N  = in_sizes[0] / DIM;   // 8192
  const int N2 = 2 * N;               // 16384

  f16*   reps    = (f16*)d_ws;                                      // 2N*256 f16 = 8 MB
  float* rowvals = (float*)((char*)d_ws + (size_t)N2 * DIM * sizeof(f16));

  // 1) normalize + fp16 pack: one wave per row, 8 waves per block
  ntxent_normalize<<<N2 / 8, 256, 0, stream>>>(zi, zj, reps, N);

  // 2) fused WMMA GEMM + masked logsumexp: 128 rows per workgroup
  ntxent_main<<<N2 / 128, 256, 0, stream>>>(reps, rowvals, N2);

  // 3) mean
  ntxent_reduce<<<1, 256, 0, stream>>>(rowvals, (float*)d_out, N2);
}
